// AFNOSpectralAttention_12429635354960
// MI455X (gfx1250) — matrix operand order repaired and checked by reading, not compiled, not run on hardware
//
#include <hip/hip_runtime.h>
#include <math.h>

// ---------------------------------------------------------------------------
// AFNO spectral attention for MI455X (gfx1250, wave32, WMMA bf16 16x16x32).
//
// x:(16,4096,768) f32. Only the first 8 W-modes are modified, so we compute
//   x_out = x + irfft2(Delta),  Delta = mixed - original on kept modes.
// Workspace (floats): Yre[S1], Yim[S1], Xout[S2];
//   S1 = 16*64*8*768 = 6,291,456  (25 MB each)
//   S2 = 16*4096*768 = 50,331,648 (201 MB)      total ~252 MB.
//
// WMMA operands are staged in LDS pre-swizzled into fragment order so each
// lane loads its 32-byte operand with 2x ds_load_b128 (no per-element gather).
//   A(16x32 bf16): lane = (m&15) + 16*((k>>3)&1), elem = (k&7) + 8*(k>>4)
//   B(32x16 bf16): lane = (n&15) + 16*((k>>3)&1), elem = (k&7) + 8*(k>>4)
// ---------------------------------------------------------------------------

#define Bn     16
#define Hn     64
#define Wn     64
#define Cn     768
#define NBn    8
#define BSn    96
#define MODESn 8
#define HWn    (Hn * Wn)

#define TWO_PI 6.283185307179586f

typedef __attribute__((ext_vector_type(16))) __bf16 v16bf;
typedef __attribute__((ext_vector_type(2)))  __bf16 bf16x2;
typedef __attribute__((ext_vector_type(8)))  float  v8f;

static __device__ __forceinline__ __bf16 tobf(float f) { return (__bf16)f; }

static __device__ __forceinline__ v8f wmma_bf16(v16bf a, v16bf b, v8f c) {
  // (neg_a, A, neg_b, B, c_mod, C, reuse_a, reuse_b) -> v_wmma_f32_16x16x32_bf16
  return __builtin_amdgcn_wmma_f32_16x16x32_bf16(false, a, false, b, (short)0, c,
                                                 false, false);
}

static __device__ __forceinline__ v16bf frag_load(const __bf16* p) {
  return *(const v16bf*)p;  // 32B aligned -> 2x ds_load_b128
}

// ---------------------------------------------------------------------------
// K1: forward DFT along W restricted to 8 modes.
//   Y[b][h][kw][c] = sum_w x[b, h*64+w, c] * e^{-2*pi*i*kw*w/64}   (unscaled)
// ---------------------------------------------------------------------------
__global__ __launch_bounds__(256)
void k_fwd_w(const float* __restrict__ x, float* __restrict__ yre,
             float* __restrict__ yim) {
  const int h = blockIdx.x, b = blockIdx.y, tid = threadIdx.x;
  __shared__ float Xs[Wn * 128];
  __shared__ float twc[MODESn * Wn], tws[MODESn * Wn];
  for (int t = tid; t < MODESn * Wn; t += 256) {
    int kw = t >> 6, w = t & 63;
    float ang = TWO_PI * (float)(kw * w) / 64.0f;
    twc[t] = cosf(ang);
    tws[t] = sinf(ang);
  }
  const long rowbase = ((long)b * HWn + (long)h * Wn) * Cn;
  for (int cc = 0; cc < Cn / 128; ++cc) {
    __syncthreads();
    for (int i = tid; i < Wn * 128; i += 256) {
      int w = i >> 7, cl = i & 127;
      Xs[i] = x[rowbase + (long)w * Cn + cc * 128 + cl];
    }
    __syncthreads();
#pragma unroll
    for (int j = 0; j < 4; ++j) {
      int o = j * 256 + tid;
      int kw = o >> 7, cl = o & 127;
      float ar = 0.f, ai = 0.f;
      for (int w = 0; w < Wn; ++w) {
        float xv = Xs[(w << 7) + cl];
        ar += xv * twc[(kw << 6) + w];
        ai -= xv * tws[(kw << 6) + w];
      }
      long oidx = (((long)(b * Hn + h) * MODESn + kw) * Cn) + cc * 128 + cl;
      yre[oidx] = ar;
      yim[oidx] = ai;
    }
  }
}

// ---------------------------------------------------------------------------
// K2/K4: dense 64-point complex DFT along H, IN PLACE.
//   forward (inverse==0): e^{-i t}, scale 1/64; inverse: e^{+i t}, scale 1.
// ---------------------------------------------------------------------------
__global__ __launch_bounds__(256)
void k_fft_h(float* __restrict__ re, float* __restrict__ im, int inverse) {
  const int kw = blockIdx.x & 7, cc = blockIdx.x >> 3, b = blockIdx.y;
  const int tid = threadIdx.x;
  __shared__ float Yr[Hn * BSn], Yi[Hn * BSn];
  __shared__ float twc[Hn], tws[Hn];
  if (tid < Hn) {
    float ang = TWO_PI * (float)tid / 64.0f;
    twc[tid] = cosf(ang);
    tws[tid] = sinf(ang);
  }
  const float sgn = inverse ? -1.f : 1.f;
  const float scale = inverse ? 1.f : (1.f / 64.f);
  for (int i = tid; i < Hn * BSn; i += 256) {
    int hh = i / BSn, c = i - hh * BSn;
    long g = (((long)(b * Hn + hh) * MODESn + kw) * Cn) + cc * BSn + c;
    Yr[i] = re[g];
    Yi[i] = im[g];
  }
  __syncthreads();
  for (int i = tid; i < Hn * BSn; i += 256) {
    int kh = i / BSn, c = i - kh * BSn;
    float zr = 0.f, zi = 0.f;
    for (int hh = 0; hh < Hn; ++hh) {
      int t = (kh * hh) & 63;
      float cv = twc[t], sv = tws[t];
      float yr = Yr[hh * BSn + c], yi = Yi[hh * BSn + c];
      zr += yr * cv + sgn * yi * sv;
      zi += yi * cv - sgn * yr * sv;
    }
    long g = (((long)(b * Hn + kh) * MODESn + kw) * Cn) + cc * BSn + c;
    re[g] = zr * scale;
    im[g] = zi * scale;
  }
}

// ---------------------------------------------------------------------------
// K3: block mixing, WMMA bf16, IN PLACE (Delta = (Z@W + bias)*gate - Z).
// Grid (mchunk=0..7, nb, b), 256 threads = 8 waves; wave = (plane, m-tile).
// M=64 rows x N=96 x K=96 per block, for real & imag planes.
// ---------------------------------------------------------------------------
__global__ __launch_bounds__(256)
void k_mix(float* __restrict__ zre, float* __restrict__ zim,
           const float* __restrict__ blockW, const float* __restrict__ blockB,
           const float* __restrict__ gates) {
  const int mchunk = blockIdx.x, nb = blockIdx.y, b = blockIdx.z;
  const int tid = threadIdx.x, lane = tid & 31, wid = tid >> 5;
  // Swizzled fragments: W[ks(3)][nt(6)] and A[plane(2)][mt(4)][ks(3)], each
  // fragment = 32 lanes x 16 bf16 = 512 elements.
  __shared__ alignas(32) __bf16 Wsw[3 * 6 * 512];     // 18 KB
  __shared__ alignas(32) __bf16 Asw[2 * 4 * 3 * 512]; // 24 KB
  __shared__ float Zr[64 * 100], Zi[64 * 100];        // f32 copy for Delta
  __shared__ float bias[BSn];

  // --- stage W (96x96) swizzled, packing bf16 pairs along K ---
  for (int p = tid; p < (BSn / 2) * BSn; p += 256) {
    int n = p % BSn, kp = p / BSn, k = kp * 2;
    float w0 = blockW[((long)nb * BSn + k) * BSn + n];
    float w1 = blockW[((long)nb * BSn + k + 1) * BSn + n];
    int ks = k >> 5, kl = k & 31;
    int ln = (n & 15) + 16 * ((kl >> 3) & 1);
    int e = (kl & 7) + 8 * (kl >> 4);
    bf16x2 pr = {tobf(w0), tobf(w1)};
    *(bf16x2*)&Wsw[((ks * 6 + (n >> 4)) * 32 + ln) * 16 + e] = pr;
  }
  if (tid < BSn) bias[tid] = blockB[nb * BSn + tid];

  // --- stage Z planes: f32 copy + swizzled bf16 copy ---
  const long rowbase = ((long)b * 512 + mchunk * 64) * Cn + nb * BSn;
#pragma unroll
  for (int plane = 0; plane < 2; ++plane) {
    const float* src = plane ? zim : zre;
    float* fdst = plane ? Zi : Zr;
    for (int p = tid; p < 64 * (BSn / 2); p += 256) {
      int m = p / 48, kp = p - (p / 48) * 48, k = kp * 2;
      float z0 = src[rowbase + (long)m * Cn + k];
      float z1 = src[rowbase + (long)m * Cn + k + 1];
      fdst[m * 100 + k] = z0;
      fdst[m * 100 + k + 1] = z1;
      int kl = k & 31;
      int ln = (m & 15) + 16 * ((kl >> 3) & 1);
      int e = (kl & 7) + 8 * (kl >> 4);
      bf16x2 pr = {tobf(z0), tobf(z1)};
      *(bf16x2*)&Asw[(((plane * 4 + (m >> 4)) * 3 + (k >> 5)) * 32 + ln) * 16 + e] = pr;
    }
  }
  const float g = 1.0f / (1.0f + expf(-gates[nb]));
  __syncthreads();

  const int mt = wid & 3, plane = wid >> 2;  // 8 waves = 2 planes x 4 m-tiles
  const int lm = lane & 15, lh = lane >> 4;
  const float* As = plane ? Zi : Zr;
  float* dst = plane ? zim : zre;
  for (int nt = 0; nt < 6; ++nt) {
    v8f acc = {};
#pragma unroll
    for (int ks = 0; ks < 3; ++ks) {
      v16bf a = frag_load(&Asw[(((plane * 4 + mt) * 3 + ks) * 32 + lane) * 16]);
      v16bf bf = frag_load(&Wsw[((ks * 6 + nt) * 32 + lane) * 16]);
      acc = wmma_bf16(a, bf, acc);
    }
    const int n = nt * 16 + lm;
#pragma unroll
    for (int r = 0; r < 8; ++r) {  // C/D layout: M = lh*8 + r, N = lm
      int mrow = mt * 16 + lh * 8 + r;
      float val = (acc[r] + bias[n]) * g;
      dst[rowbase + (long)mrow * Cn + n] = val - As[mrow * 100 + n];
    }
  }
}

// ---------------------------------------------------------------------------
// K5: 8-mode irfft along W (Hermitian weights: 1 for kw=0, else 2; x 1/64
// ortho), fused residual add:  x_out = x + Re{ sum_kw wt*U*e^{+i.}}.
// ---------------------------------------------------------------------------
__global__ __launch_bounds__(256)
void k_inv_w(const float* __restrict__ ure, const float* __restrict__ uim,
             const float* __restrict__ x, float* __restrict__ xout) {
  const int h = blockIdx.x, b = blockIdx.y, tid = threadIdx.x;
  __shared__ float Ur[MODESn * Cn], Ui[MODESn * Cn];
  __shared__ float twc[MODESn * Wn], tws[MODESn * Wn];
  for (int t = tid; t < MODESn * Wn; t += 256) {
    int kw = t >> 6, w = t & 63;
    float wt = (kw == 0 ? 1.0f : 2.0f) / 64.0f;
    float ang = TWO_PI * (float)(kw * w) / 64.0f;
    twc[t] = wt * cosf(ang);
    tws[t] = wt * sinf(ang);
  }
  const long ubase = (long)(b * Hn + h) * MODESn * Cn;
  for (int i = tid; i < MODESn * Cn; i += 256) {
    Ur[i] = ure[ubase + i];
    Ui[i] = uim[ubase + i];
  }
  __syncthreads();
  const long rowbase = ((long)b * HWn + (long)h * Wn) * Cn;
  for (int i = tid; i < Wn * Cn; i += 256) {
    int w = i / Cn, c = i - w * Cn;
    float s = 0.f;
#pragma unroll
    for (int kw = 0; kw < MODESn; ++kw)
      s += Ur[kw * Cn + c] * twc[(kw << 6) + w] -
           Ui[kw * Cn + c] * tws[(kw << 6) + w];
    xout[rowbase + i] = x[rowbase + i] + s;
  }
}

// ---------------------------------------------------------------------------
// K6: projection GEMM (65536x768x768) + residual.
//   out[m,n] = x_out[m,n] + rescale*(sum_k x_out[m,k]*W_out[n,k] + b_out[n])
// Block tile 64x128, K stepped by 32; 8 waves: 4 m-tiles x 2 n-groups of 4.
// ---------------------------------------------------------------------------
__global__ __launch_bounds__(256)
void k_proj(const float* __restrict__ xo, const float* __restrict__ wout,
            const float* __restrict__ bout, const float* __restrict__ rescale_p,
            float* __restrict__ out) {
  const int bm = blockIdx.x, bn = blockIdx.y;
  const int tid = threadIdx.x, lane = tid & 31, wid = tid >> 5;
  __shared__ alignas(32) __bf16 Asw[4 * 512];  // 4 m-frags (64x32)   4 KB
  __shared__ alignas(32) __bf16 Bsw[8 * 512];  // 8 n-frags (32x128)  8 KB
  const int m0 = bm * 64, n0 = bn * 128;
  const int mt = wid >> 1, ng = wid & 1;
  const int lm = lane & 15, lh = lane >> 4;
  v8f acc[4] = {};
  for (int kt = 0; kt < Cn / 32; ++kt) {
    __syncthreads();
    const int k0 = kt * 32;
    // A: 64 rows x 32 k -> 1024 bf16 pairs (pairs along K)
#pragma unroll
    for (int i = 0; i < 4; ++i) {
      int p = i * 256 + tid;
      int ml = p >> 4, k = (p & 15) * 2;
      float a0 = xo[(long)(m0 + ml) * Cn + k0 + k];
      float a1 = xo[(long)(m0 + ml) * Cn + k0 + k + 1];
      int ln = (ml & 15) + 16 * ((k >> 3) & 1);
      int e = (k & 7) + 8 * (k >> 4);
      bf16x2 pr = {tobf(a0), tobf(a1)};
      *(bf16x2*)&Asw[(((ml >> 4) * 32 + ln) * 16) + e] = pr;
    }
    // B: 32 k x 128 n -> 2048 bf16 pairs (pairs along K)
#pragma unroll
    for (int i = 0; i < 8; ++i) {
      int p = i * 256 + tid;
      int nl = p >> 4, k = (p & 15) * 2;
      float b0 = wout[(long)(n0 + nl) * Cn + k0 + k];
      float b1 = wout[(long)(n0 + nl) * Cn + k0 + k + 1];
      int ln = (nl & 15) + 16 * ((k >> 3) & 1);
      int e = (k & 7) + 8 * (k >> 4);
      bf16x2 pr = {tobf(b0), tobf(b1)};
      *(bf16x2*)&Bsw[(((nl >> 4) * 32 + ln) * 16) + e] = pr;
    }
    if (kt + 1 < Cn / 32)  // CDNA5 global_prefetch_b8 for next A tile
      __builtin_prefetch(&xo[(long)(m0 + (tid >> 5)) * Cn + k0 + 32 + (tid & 31)],
                         0, 1);
    __syncthreads();
    v16bf a = frag_load(&Asw[(mt * 32 + lane) * 16]);
#pragma unroll
    for (int t = 0; t < 4; ++t) {
      v16bf bf = frag_load(&Bsw[(((ng * 4 + t) * 32) + lane) * 16]);
      acc[t] = wmma_bf16(a, bf, acc[t]);
    }
  }
  const float rs = rescale_p[0];
#pragma unroll
  for (int t = 0; t < 4; ++t) {
    int n = n0 + (ng * 4 + t) * 16 + lm;
    float bo = bout[n];
#pragma unroll
    for (int r = 0; r < 8; ++r) {
      int m = m0 + mt * 16 + lh * 8 + r;
      long gi = (long)m * Cn + n;
      float xv = xo[gi];
      out[gi] = xv + rs * (acc[t][r] + bo);
    }
  }
}

// ---------------------------------------------------------------------------
extern "C" void kernel_launch(void* const* d_in, const int* in_sizes, int n_in,
                              void* d_out, int out_size, void* d_ws,
                              size_t ws_size, hipStream_t stream) {
  const float* x       = (const float*)d_in[0];
  const float* blockW  = (const float*)d_in[1];
  const float* blockB  = (const float*)d_in[2];
  const float* gates   = (const float*)d_in[3];
  const float* wout    = (const float*)d_in[4];
  const float* bout    = (const float*)d_in[5];
  const float* rescale = (const float*)d_in[6];
  float* out = (float*)d_out;

  const size_t S1 = (size_t)Bn * Hn * MODESn * Cn;  // 6,291,456 floats
  float* Yre  = (float*)d_ws;        // spectral real plane (reused in place)
  float* Yim  = Yre + S1;            // spectral imag plane (reused in place)
  float* Xout = Yim + S1;            // x_out, (B*HW, C)

  k_fwd_w<<<dim3(Hn, Bn), 256, 0, stream>>>(x, Yre, Yim);
  k_fft_h<<<dim3(64, Bn), 256, 0, stream>>>(Yre, Yim, 0);
  k_mix  <<<dim3(8, NBn, Bn), 256, 0, stream>>>(Yre, Yim, blockW, blockB, gates);
  k_fft_h<<<dim3(64, Bn), 256, 0, stream>>>(Yre, Yim, 1);
  k_inv_w<<<dim3(Hn, Bn), 256, 0, stream>>>(Yre, Yim, x, Xout);
  k_proj <<<dim3((Bn * HWn) / 64, Cn / 128), 256, 0, stream>>>(Xout, wout, bout,
                                                              rescale, out);
}
